// MultiHeadAttentionWithRelativePosition_65670049956501
// MI455X (gfx1250) — matrix operand order repaired and checked by reading, not compiled, not run on hardware
//
#include <hip/hip_runtime.h>
#include <hip/hip_bf16.h>

// MI455X / gfx1250, wave32. bf16 WMMA (16x16x32, fp32 accum) everywhere.
typedef __attribute__((ext_vector_type(16))) __bf16 v16bf;
typedef __attribute__((ext_vector_type(8)))  float  v8f;
typedef int v4i_vs __attribute__((vector_size(16)));   // matches async-LDS builtin param

#define WMMA_BF16(a, b, c) \
  __builtin_amdgcn_wmma_f32_16x16x32_bf16(false, (a), false, (b), (short)0, (c), false, false)

#if __has_builtin(__builtin_amdgcn_global_load_async_to_lds_b128) && \
    __has_builtin(__builtin_amdgcn_s_wait_asynccnt)
#define HAS_ASYNC_LDS 1
#else
#define HAS_ASYNC_LDS 0
#endif

#if HAS_ASYNC_LDS
#define AS_GLOBAL(p) ((__attribute__((address_space(1))) v4i_vs*)(p))
#define AS_LOCAL(p)  ((__attribute__((address_space(3))) v4i_vs*)(p))
#endif

static constexpr int Bsz = 2, S = 2048, Cdim = 1024, H = 16, Dh = 64;
static constexpr int M_ROWS = Bsz * S;          // 4096
static constexpr int TBL_ROWS = 2 * S - 1;      // 4095
static constexpr int TBL_ROWS_PAD = 4160;       // padded so banded tiles never clamp

// ---------------------------------------------------------------------------
// Kernel 0: rel_k_table fp32 -> bf16 once (0.5 MB, stays L2-resident), padded.
// ---------------------------------------------------------------------------
__global__ void table_bf16_kernel(const float* __restrict__ t, __bf16* __restrict__ o) {
  const int i = blockIdx.x * blockDim.x + threadIdx.x;
  if (i < TBL_ROWS_PAD * Dh)
    o[i] = (i < TBL_ROWS * Dh) ? (__bf16)t[i] : (__bf16)0.0f;
}

// ---------------------------------------------------------------------------
// Kernel 1: q/k/v = src @ W^T + b, output stored as bf16 (flat (B,S,C) layout)
// 64x64 output tile per block, 4 waves, K-loop in steps of 32, bf16 in LDS.
// ---------------------------------------------------------------------------
__global__ void qkv_proj_kernel(const float* __restrict__ src,
                                const float* __restrict__ Wq, const float* __restrict__ bq,
                                const float* __restrict__ Wk, const float* __restrict__ bk,
                                const float* __restrict__ Wv, const float* __restrict__ bv,
                                __bf16* __restrict__ qo, __bf16* __restrict__ ko,
                                __bf16* __restrict__ vo) {
  __shared__ __bf16 lsA[64 * 32];
  __shared__ __bf16 lsB[64 * 32];

  const float* W;
  const float* bias;
  __bf16* dst;
  if (blockIdx.z == 0)      { W = Wq; bias = bq; dst = qo; }
  else if (blockIdx.z == 1) { W = Wk; bias = bk; dst = ko; }
  else                      { W = Wv; bias = bv; dst = vo; }

  const int m0   = blockIdx.x * 64;
  const int n0   = blockIdx.y * 64;
  const int tid  = threadIdx.x;
  const int wid  = tid >> 5;
  const int lane = tid & 31;
  const int l15  = lane & 15;
  const int hi   = lane >> 4;
  const int kb8  = hi * 8;

  v8f zero = {};
  v8f acc[4];
  for (int j = 0; j < 4; ++j) acc[j] = zero;

  for (int kk = 0; kk < Cdim; kk += 32) {
    for (int i = tid; i < 64 * 32; i += 128) {
      int r = i >> 5, c = i & 31;
      lsA[i] = (__bf16)src[(size_t)(m0 + r) * Cdim + kk + c];
      lsB[i] = (__bf16)W[(size_t)(n0 + r) * Cdim + kk + c];
    }
    __syncthreads();

    v16bf a;  // A layout: row = wid*16+l15; lanes<16 K 0..7/16..23, lanes>=16 K 8..15/24..31
    const int arow = wid * 16 + l15;
#pragma unroll
    for (int i = 0; i < 8; ++i) {
      a[i]     = lsA[arow * 32 + kb8 + i];
      a[i + 8] = lsA[arow * 32 + 16 + kb8 + i];
    }
#pragma unroll
    for (int j = 0; j < 4; ++j) {
      v16bf bb;  // B[k,n]: n = j*16+l15 (row of lsB), K = hi*8 + e + (e & 8)
      const int brow = j * 16 + l15;
#pragma unroll
      for (int e = 0; e < 16; ++e) bb[e] = lsB[brow * 32 + hi * 8 + e + (e & 8)];
      acc[j] = WMMA_BF16(a, bb, acc[j]);
    }
    __syncthreads();
  }

#pragma unroll
  for (int j = 0; j < 4; ++j)
#pragma unroll
    for (int r = 0; r < 8; ++r) {
      const int m = m0 + wid * 16 + r + 8 * hi;
      const int n = n0 + j * 16 + l15;
      dst[(size_t)m * Cdim + n] = (__bf16)(acc[j][r] + bias[n]);
    }
}

// ---------------------------------------------------------------------------
// Kernel 2: fused attention, flash-style online softmax.
// Grid (S/128, B*H). 256 threads = 8 waves; wave w fully owns query rows
// s0 + 16w .. +16 (content + rel-pos band GEMM + softmax + PV).
// K/V/table tiles DMA'd global->LDS via GLOBAL_LOAD_ASYNC_TO_LDS_B128 and
// synchronized with s_wait_asynccnt (fallback: VALU staging loop).
// Dynamic LDS layout (128 KB):
//   [0      ,  8192) K tile   64x64 bf16
//   [8192   , 16384) V tile   64x64 bf16
//   [16384  , 49152) table    256x64 bf16 (banded rel_k rows)
//   [49152  ,114688) Ppos     8 x 16x128 f32 (per-wave rel-pos GEMM result)
//   [114688 ,131072) Pprob    8 x 16x64  bf16 (per-wave probabilities)
// ---------------------------------------------------------------------------
__global__ void attn_relpos_kernel(const __bf16* __restrict__ q,
                                   const __bf16* __restrict__ k,
                                   const __bf16* __restrict__ v,
                                   const __bf16* __restrict__ tbl,
                                   float* __restrict__ out) {
  extern __shared__ char smem[];
  __bf16* lsK = (__bf16*)(smem);
  __bf16* lsV = (__bf16*)(smem + 8192);
  __bf16* lsT = (__bf16*)(smem + 16384);
  float*  lsP = (float*)(smem + 49152);
  __bf16* lsA = (__bf16*)(smem + 114688);

  const int tid  = threadIdx.x;
  const int wid  = tid >> 5;
  const int lane = tid & 31;
  const int l15  = lane & 15;
  const int hi   = lane >> 4;
  const int kb8  = hi * 8;

  const int bh = blockIdx.y;        // 0..31
  const int b  = bh >> 4;
  const int h  = bh & 15;
  const int s0  = blockIdx.x * 128;
  const int s0w = s0 + wid * 16;
  const int offw = 112 - wid * 16;  // per-wave offset of its j-band inside lsT

  const __bf16* qslab = q + (size_t)b * (S * Cdim) + (size_t)h * (S * Dh);
  const __bf16* kslab = k + (size_t)b * (S * Cdim) + (size_t)h * (S * Dh);
  const __bf16* vslab = v + (size_t)b * (S * Cdim) + (size_t)h * (S * Dh);
  const __bf16* qrp   = q + (size_t)bh * Dh;  // q_r[s,bh,d] = q_flat[s*2048 + bh*64 + d]

  // A fragments for Q (content) and q_r (rel-pos), held across the whole loop
  const int myrow = s0w + l15;
  v16bf aq[2], ar[2];
#pragma unroll
  for (int ks = 0; ks < 2; ++ks)
#pragma unroll
    for (int i = 0; i < 8; ++i) {
      aq[ks][i]     = qslab[(size_t)myrow * Dh + ks * 32 + kb8 + i];
      aq[ks][i + 8] = qslab[(size_t)myrow * Dh + ks * 32 + 16 + kb8 + i];
      ar[ks][i]     = qrp[(size_t)myrow * 2048 + ks * 32 + kb8 + i];
      ar[ks][i + 8] = qrp[(size_t)myrow * 2048 + ks * 32 + 16 + kb8 + i];
    }

  v8f zero = {};
  v8f O[4];
  for (int j = 0; j < 4; ++j) O[j] = zero;
  float mrow[8], lrow[8];
#pragma unroll
  for (int r = 0; r < 8; ++r) { mrow[r] = -1e30f; lrow[r] = 0.f; }

  for (int t0 = 0; t0 < S; t0 += 64) {
    const int jblk = t0 - s0 + 1920;  // >= 0; padded table makes jblk+255 safe

#if HAS_ASYNC_LDS
    {  // DMA tiles straight into LDS; each of the 3 regions is contiguous bytes
      const char* gk = (const char*)(kslab + (size_t)t0 * Dh);
      const char* gv = (const char*)(vslab + (size_t)t0 * Dh);
      const char* gt = (const char*)(tbl + (size_t)jblk * Dh);
#pragma unroll
      for (int c = 0; c < 2; ++c) {  // K,V: 8192 B each = 2 x (256 lanes x 16 B)
        const int off = (tid + c * 256) * 16;
        __builtin_amdgcn_global_load_async_to_lds_b128(AS_GLOBAL(gk + off),
                                                       AS_LOCAL((char*)lsK + off), 0, 0);
        __builtin_amdgcn_global_load_async_to_lds_b128(AS_GLOBAL(gv + off),
                                                       AS_LOCAL((char*)lsV + off), 0, 0);
      }
#pragma unroll
      for (int c = 0; c < 8; ++c) {  // table band: 32768 B = 8 x (256 x 16 B)
        const int off = (tid + c * 256) * 16;
        __builtin_amdgcn_global_load_async_to_lds_b128(AS_GLOBAL(gt + off),
                                                       AS_LOCAL((char*)lsT + off), 0, 0);
      }
      __builtin_amdgcn_s_wait_asynccnt(0);
    }
#else
    for (int i = tid; i < 64 * 64; i += 256) {
      lsK[i] = kslab[(size_t)t0 * Dh + i];
      lsV[i] = vslab[(size_t)t0 * Dh + i];
    }
    for (int i = tid; i < 256 * 64; i += 256) lsT[i] = tbl[(size_t)jblk * Dh + i];
#endif
    if (t0 + 64 < S) {  // prefetch next K/V tiles into L2 (global_prefetch_b8)
      __builtin_prefetch(kslab + (size_t)(t0 + 64 + (lane & 7) * 8) * Dh, 0, 0);
      __builtin_prefetch(vslab + (size_t)(t0 + 64 + (lane & 7) * 8) * Dh, 0, 0);
    }
    __syncthreads();

    // ---- content scores: Q(16x64) @ K_tile^T -> 4 subtiles of 16x16
    v8f sc[4];
#pragma unroll
    for (int j = 0; j < 4; ++j) {
      v8f acc = zero;
#pragma unroll
      for (int ks = 0; ks < 2; ++ks) {
        v16bf bk_;
        const int trow = j * 16 + l15;
        const int base = trow * 64 + ks * 32 + hi * 8;
#pragma unroll
        for (int e = 0; e < 16; ++e) bk_[e] = lsK[base + e + (e & 8)];
        acc = WMMA_BF16(aq[ks], bk_, acc);
      }
      sc[j] = acc;
    }

    // ---- rel-pos band GEMM: q_r(16x64) @ table^T(64x128) -> lsP[wid]
#pragma unroll
    for (int jj = 0; jj < 8; ++jj) {
      v8f acc = zero;
#pragma unroll
      for (int ks = 0; ks < 2; ++ks) {
        v16bf bt;
        const int trow = offw + jj * 16 + l15;
        const int base = trow * 64 + ks * 32 + hi * 8;
#pragma unroll
        for (int e = 0; e < 16; ++e) bt[e] = lsT[base + e + (e & 8)];
        acc = WMMA_BF16(ar[ks], bt, acc);
      }
#pragma unroll
      for (int r = 0; r < 8; ++r) {
        const int m = r + 8 * hi;
        lsP[(wid * 16 + m) * 128 + jj * 16 + l15] = acc[r];
      }
    }
    // same-wave LDS RAW: DS ops are in-order within a wave, no barrier needed

    // ---- gather diagonal band, combine, online softmax (rows owned per-wave)
    float pv[4][8];
    float rmax[8];
#pragma unroll
    for (int r = 0; r < 8; ++r) rmax[r] = -1e30f;
#pragma unroll
    for (int j = 0; j < 4; ++j)
#pragma unroll
      for (int r = 0; r < 8; ++r) {
        const int m = r + 8 * hi;
        const int n = j * 16 + l15;
        const float pos = lsP[(wid * 16 + m) * 128 + (n - m + 15)];
        const float sv = (sc[j][r] + pos) * 0.125f;  // 1/sqrt(64)
        pv[j][r] = sv;
        rmax[r] = fmaxf(rmax[r], sv);
      }
#pragma unroll
    for (int d2 = 1; d2 < 16; d2 <<= 1)
#pragma unroll
      for (int r = 0; r < 8; ++r)
        rmax[r] = fmaxf(rmax[r], __shfl_xor(rmax[r], d2, 32));

    float scl[8];
#pragma unroll
    for (int r = 0; r < 8; ++r) {
      const float mn = fmaxf(mrow[r], rmax[r]);
      scl[r] = __expf(mrow[r] - mn);
      mrow[r] = mn;
    }
    float rsum[8];
#pragma unroll
    for (int r = 0; r < 8; ++r) rsum[r] = 0.f;
#pragma unroll
    for (int j = 0; j < 4; ++j)
#pragma unroll
      for (int r = 0; r < 8; ++r) {
        const float p = __expf(pv[j][r] - mrow[r]);
        pv[j][r] = p;
        rsum[r] += p;
      }
#pragma unroll
    for (int d2 = 1; d2 < 16; d2 <<= 1)
#pragma unroll
      for (int r = 0; r < 8; ++r) rsum[r] += __shfl_xor(rsum[r], d2, 32);
#pragma unroll
    for (int r = 0; r < 8; ++r) lrow[r] = lrow[r] * scl[r] + rsum[r];
#pragma unroll
    for (int j = 0; j < 4; ++j)
#pragma unroll
      for (int r = 0; r < 8; ++r) O[j][r] *= scl[r];

    // ---- probabilities -> LDS (C-layout -> A-layout redistribution)
#pragma unroll
    for (int j = 0; j < 4; ++j)
#pragma unroll
      for (int r = 0; r < 8; ++r) {
        const int m = r + 8 * hi;
        lsA[(wid * 16 + m) * 64 + j * 16 + l15] = (__bf16)pv[j][r];
      }

    // ---- O += P(16x64) @ V_tile(64x64)
#pragma unroll
    for (int ks = 0; ks < 2; ++ks) {
      v16bf ap;
      const int arow = wid * 16 + l15;
#pragma unroll
      for (int i = 0; i < 8; ++i) {
        ap[i]     = lsA[arow * 64 + ks * 32 + kb8 + i];
        ap[i + 8] = lsA[arow * 64 + ks * 32 + 16 + kb8 + i];
      }
#pragma unroll
      for (int j = 0; j < 4; ++j) {
        v16bf bv_;  // B[k=t, n=d]: column reads of lsV
#pragma unroll
        for (int e = 0; e < 16; ++e) {
          const int trow = ks * 32 + hi * 8 + e + (e & 8);
          bv_[e] = lsV[trow * 64 + j * 16 + l15];
        }
        O[j] = WMMA_BF16(ap, bv_, O[j]);
      }
    }
    __syncthreads();  // protect lsK/lsV/lsT for next iteration
  }

  // ---- epilogue: normalize, write out[b, s, h*64 + d] (fp32)
#pragma unroll
  for (int j = 0; j < 4; ++j)
#pragma unroll
    for (int r = 0; r < 8; ++r) {
      const int m = r + 8 * hi;
      const int srow = s0w + m;
      const float val = O[j][r] / lrow[r];
      out[(size_t)b * S * Cdim + (size_t)srow * Cdim + h * 64 + j * 16 + l15] = val;
    }
}

// ---------------------------------------------------------------------------
extern "C" void kernel_launch(void* const* d_in, const int* in_sizes, int n_in,
                              void* d_out, int out_size, void* d_ws, size_t ws_size,
                              hipStream_t stream) {
  const float* src = (const float*)d_in[0];
  const float* Wq  = (const float*)d_in[1];
  const float* bq  = (const float*)d_in[2];
  const float* Wk  = (const float*)d_in[3];
  const float* bk  = (const float*)d_in[4];
  const float* Wv  = (const float*)d_in[5];
  const float* bv  = (const float*)d_in[6];
  const float* tbl = (const float*)d_in[7];

  // workspace layout: padded bf16 table first (so band over-reads stay inside),
  // then bf16 q/k/v.
  __bf16* tblb = (__bf16*)d_ws;                        // 4160*64 = 266,240 elems
  __bf16* qb   = tblb + (size_t)TBL_ROWS_PAD * Dh;
  __bf16* kb   = qb + (size_t)M_ROWS * Cdim;
  __bf16* vb   = kb + (size_t)M_ROWS * Cdim;

  table_bf16_kernel<<<(TBL_ROWS_PAD * Dh + 255) / 256, 256, 0, stream>>>(tbl, tblb);

  dim3 g1(M_ROWS / 64, Cdim / 64, 3);   // 64 x 16 x 3
  qkv_proj_kernel<<<g1, 128, 0, stream>>>(src, Wq, bq, Wk, bk, Wv, bv, qb, kb, vb);

  dim3 g2(S / 128, Bsz * H);            // 16 x 32
  const size_t shmem = 131072;          // 128 KB of the 320 KB/WGP LDS
  attn_relpos_kernel<<<g2, 256, shmem, stream>>>(qb, kb, vb, tblb, (float*)d_out);
}
